// EncoderLayer_31275951849624
// MI455X (gfx1250) — matrix operand order repaired and checked by reading, not compiled, not run on hardware
//
#include <hip/hip_runtime.h>
#include <math.h>

// ---------------------------------------------------------------------------
// Types for CDNA5 WMMA (gfx1250, wave32)
// ---------------------------------------------------------------------------
typedef __bf16 bf16_t;
typedef bf16_t v16bf __attribute__((ext_vector_type(16)));
typedef bf16_t v8bf  __attribute__((ext_vector_type(8)));
typedef float  v8f   __attribute__((ext_vector_type(8)));
typedef int    v4i   __attribute__((ext_vector_type(4)));

#define R_ROWS   8192      // BATCH * SEQLEN
#define D_MODEL  1024
#define D_INNER  2048
#define D_STATE  16
#define DT_RANK  64
#define SEQLEN_C 2048
#define BATCH_C  4
#define PROJ_LD  128       // proj padded from 96 -> 128 cols

// gfx1250 async global->LDS staging (ASYNCcnt path); sync fallback otherwise.
// Probe-confirmed signature: (as1 int4* src, as3 int4* dst, imm off, imm cpol)
#if __has_builtin(__builtin_amdgcn_global_load_async_to_lds_b128)
  #define ASYNC_LDS 1
  typedef __attribute__((address_space(1))) v4i* gvp_t;
  typedef __attribute__((address_space(3))) v4i* lvp_t;
#else
  #define ASYNC_LDS 0
#endif

#if __has_builtin(__builtin_amdgcn_s_wait_asynccnt)
  #define WAIT_ASYNCCNT0() __builtin_amdgcn_s_wait_asynccnt(0)
#else
  #define WAIT_ASYNCCNT0() asm volatile("s_wait_asynccnt 0x0" ::: "memory")
#endif

// ---------------------------------------------------------------------------
// bf16 tiled GEMM:  C[M,N] (f32) = A[M,K] (bf16 row-major) * Bt[N,K] (bf16)
//   Bt is B pre-transposed (and zero-padded to N%128==0) so both operands
//   stream K-contiguous and NO bounds checks are needed anywhere.
//   Block: 256 threads = 8 waves. Block tile 128x128, wave tile 64x32
//   (4x2 subtiles of 16x16), K step 32 -> v_wmma_f32_16x16x32_bf16.
//   Double-buffered LDS; next tile staged with GLOBAL_LOAD_ASYNC_TO_LDS_B128
//   (ASYNCcnt) while the current tile feeds the matrix cores.
//   Requires: M % 128 == 0, N % 128 == 0, K % 32 == 0.
// ---------------------------------------------------------------------------
__global__ __launch_bounds__(256)
void k_gemm_bf16(const bf16_t* __restrict__ A, const bf16_t* __restrict__ Bt,
                 float* __restrict__ C, int M, int N, int K) {
    __shared__ bf16_t As[2][128][40];   // +8 pad, 80B rows (16B aligned)
    __shared__ bf16_t Bs[2][128][40];

    const int tid  = threadIdx.x;
    const int lane = tid & 31;
    const int wave = tid >> 5;
    const int wm   = wave >> 2;            // 0..1  (M half)
    const int wn   = wave & 3;             // 0..3  (N quarter)
    const int g    = lane >> 4;            // lane group (K-half select)
    const int r16  = lane & 15;

    const size_t blockM = (size_t)blockIdx.y * 128;
    const size_t blockN = (size_t)blockIdx.x * 128;

    // staging assignment: each thread moves 4 x 16B per tile pair
    const int r0 = tid >> 2;               // 0..63
    const int c0 = (tid & 3) << 3;         // 0,8,16,24
    const int r1 = r0 + 64;                // 64..127

    const bf16_t* gA0 = A  + (blockM + (size_t)r0) * K + c0;
    const bf16_t* gA1 = A  + (blockM + (size_t)r1) * K + c0;
    const bf16_t* gB0 = Bt + (blockN + (size_t)r0) * K + c0;
    const bf16_t* gB1 = Bt + (blockN + (size_t)r1) * K + c0;

    auto stage = [&](int b) {
#if ASYNC_LDS
        __builtin_amdgcn_global_load_async_to_lds_b128(
            (gvp_t)gA0, (lvp_t)&As[b][r0][c0], 0, 0);
        __builtin_amdgcn_global_load_async_to_lds_b128(
            (gvp_t)gA1, (lvp_t)&As[b][r1][c0], 0, 0);
        __builtin_amdgcn_global_load_async_to_lds_b128(
            (gvp_t)gB0, (lvp_t)&Bs[b][r0][c0], 0, 0);
        __builtin_amdgcn_global_load_async_to_lds_b128(
            (gvp_t)gB1, (lvp_t)&Bs[b][r1][c0], 0, 0);
#else
        *(uint4*)&As[b][r0][c0] = *(const uint4*)gA0;
        *(uint4*)&As[b][r1][c0] = *(const uint4*)gA1;
        *(uint4*)&Bs[b][r0][c0] = *(const uint4*)gB0;
        *(uint4*)&Bs[b][r1][c0] = *(const uint4*)gB1;
#endif
        gA0 += 32; gA1 += 32; gB0 += 32; gB1 += 32;
    };

    v8f acc[4][2];
    const v8f vzero = {};
    #pragma unroll
    for (int i = 0; i < 4; ++i)
        #pragma unroll
        for (int j = 0; j < 2; ++j) acc[i][j] = vzero;

    const int nsteps = K >> 5;
    stage(0);
#if ASYNC_LDS
    WAIT_ASYNCCNT0();
#endif
    __syncthreads();

    for (int s = 0; s < nsteps; ++s) {
        const int cur = s & 1;
        if (s + 1 < nsteps) stage(1 - cur);

        union Frag { v16bf v; v8bf h[2]; };
        Frag af[4], bfr[2];
        // A fragment (16x32 bf16): lane g selects K {8g..8g+7, 16+8g..16+8g+7}
        #pragma unroll
        for (int i = 0; i < 4; ++i) {
            const int row = wm * 64 + i * 16 + r16;
            af[i].h[0] = *(const v8bf*)&As[cur][row][8 * g];
            af[i].h[1] = *(const v8bf*)&As[cur][row][16 + 8 * g];
        }
        // B fragment (32x16 bf16): lane g selects K {8g..8g+7, 16+8g..16+8g+7}
        #pragma unroll
        for (int j = 0; j < 2; ++j) {
            const int col = wn * 32 + j * 16 + r16;
            bfr[j].h[0] = *(const v8bf*)&Bs[cur][col][8 * g];
            bfr[j].h[1] = *(const v8bf*)&Bs[cur][col][16 + 8 * g];
        }
        #pragma unroll
        for (int i = 0; i < 4; ++i)
            #pragma unroll
            for (int j = 0; j < 2; ++j)
                acc[i][j] = __builtin_amdgcn_wmma_f32_16x16x32_bf16(
                    false, af[i].v, false, bfr[j].v,
                    (short)0, acc[i][j], false, false);
#if ASYNC_LDS
        if (s + 1 < nsteps) WAIT_ASYNCCNT0();
#endif
        __syncthreads();
    }

    // C/D layout: VGPR v -> M = v + 8*g, lane[3:0] -> N.  N%128==0: no guards.
    #pragma unroll
    for (int i = 0; i < 4; ++i)
        #pragma unroll
        for (int j = 0; j < 2; ++j) {
            const int col = (int)blockN + wn * 32 + j * 16 + r16;
            const size_t rbase = blockM + (size_t)(wm * 64 + i * 16 + 8 * g);
            #pragma unroll
            for (int v = 0; v < 8; ++v)
                C[(rbase + v) * N + col] = acc[i][j][v];
        }
}

// ---------------------------------------------------------------------------
// Weight cast + transpose + row pad: w[K,N] f32 -> wT[Npad,K] bf16 (zeros pad)
// ---------------------------------------------------------------------------
__global__ void k_cast_transpose(const float* __restrict__ w,
                                 bf16_t* __restrict__ wT, int K, int N, int Npad) {
    const int idx = blockIdx.x * 256 + threadIdx.x;
    if (idx >= K * Npad) return;
    const int n = idx / K;
    const int k = idx - n * K;
    wT[(size_t)n * K + k] = (n < N) ? (bf16_t)w[(size_t)k * N + n] : (bf16_t)0.f;
}

// RMSNorm row kernel: x[row,1024] f32 -> xn bf16
__global__ __launch_bounds__(256)
void k_rmsnorm(const float* __restrict__ x, const float* __restrict__ w,
               bf16_t* __restrict__ outp) {
    const int row = blockIdx.x;
    const int C = D_MODEL;
    const float* xr = x + (size_t)row * C;
    __shared__ float red[256];
    float s = 0.f;
    for (int c = threadIdx.x; c < C; c += 256) { const float v = xr[c]; s += v * v; }
    red[threadIdx.x] = s;
    __syncthreads();
    for (int st = 128; st > 0; st >>= 1) {
        if (threadIdx.x < st) red[threadIdx.x] += red[threadIdx.x + st];
        __syncthreads();
    }
    const float inv = rsqrtf(red[0] / (float)C + 1e-6f);
    for (int c = threadIdx.x; c < C; c += 256)
        outp[(size_t)row * C + c] = (bf16_t)(xr[c] * inv * w[c]);
}

// Depthwise causal conv (k=4) + bias + SiLU over xz[:, :2048]
__global__ __launch_bounds__(256)
void k_conv_silu(const float* __restrict__ xz, const float* __restrict__ cw,
                 const float* __restrict__ cb, float* __restrict__ xi,
                 bf16_t* __restrict__ xi_bf) {
    const size_t idx = (size_t)blockIdx.x * 256 + threadIdx.x;  // over R*2048
    const int d = (int)(idx & (D_INNER - 1));
    const size_t row = idx >> 11;                 // /2048
    const int b = (int)(row >> 11);               // row / 2048
    const int t = (int)(row & (SEQLEN_C - 1));
    float acc = cb[d];
    #pragma unroll
    for (int j = 0; j < 4; ++j) {
        const int tt = t - 3 + j;
        if (tt >= 0)
            acc += cw[d * 4 + j] *
                   xz[((size_t)b * SEQLEN_C + tt) * (2 * D_INNER) + d];
    }
    const float s = acc / (1.f + __expf(-acc));
    xi[idx] = s;
    xi_bf[idx] = (bf16_t)s;
}

// Slice f32 [rows, ld] cols [col0, col0+cols) -> bf16 [rows, cols]
__global__ void k_slice_cast_bf16(const float* __restrict__ in,
                                  bf16_t* __restrict__ outp,
                                  int rows, int ld, int col0, int cols) {
    const int idx = blockIdx.x * 256 + threadIdx.x;
    if (idx >= rows * cols) return;
    const int r = idx / cols;
    const int c = idx - r * cols;
    outp[idx] = (bf16_t)in[(size_t)r * ld + col0 + c];
}

// dt = softplus(dt_raw + dt_b) in-place
__global__ void k_softplus(float* __restrict__ dt, const float* __restrict__ dt_b,
                           int total) {
    const int idx = blockIdx.x * 256 + threadIdx.x;
    if (idx >= total) return;
    const int d = idx & (D_INNER - 1);
    const float v = dt[idx] + dt_b[d];
    dt[idx] = (v > 20.f) ? v : log1pf(__expf(v));
}

// Selective scan: per (b,d) channel, 16-state recurrence in registers.
// B/C staged in LDS 16 timesteps at a time (2 barriers per 16 steps).
__global__ __launch_bounds__(256)
void k_scan(const float* __restrict__ xi, const float* __restrict__ dt,
            const float* __restrict__ proj, const float* __restrict__ A_log,
            float* __restrict__ ys) {
    const int b = blockIdx.y;
    const int d = blockIdx.x * 256 + threadIdx.x;   // 0..2047
    float A[D_STATE], h[D_STATE];
    #pragma unroll
    for (int n = 0; n < D_STATE; ++n) {
        A[n] = -__expf(A_log[(size_t)d * D_STATE + n]);
        h[n] = 0.f;
    }
    __shared__ float BCb[16][32];
    for (int t0 = 0; t0 < SEQLEN_C; t0 += 16) {
        __syncthreads();
        #pragma unroll
        for (int c = threadIdx.x; c < 512; c += 256) {
            const int tt = c >> 5, n = c & 31;
            BCb[tt][n] = proj[((size_t)b * SEQLEN_C + t0 + tt) * PROJ_LD
                              + DT_RANK + n];
        }
        __syncthreads();
        #pragma unroll
        for (int j = 0; j < 16; ++j) {
            const size_t row = (size_t)b * SEQLEN_C + t0 + j;
            const float xt  = xi[row * D_INNER + d];
            const float dtt = dt[row * D_INNER + d];
            const float dx  = dtt * xt;
            float y = 0.f;
            #pragma unroll
            for (int n = 0; n < D_STATE; ++n) {
                h[n] = __expf(dtt * A[n]) * h[n] + dx * BCb[j][n];
                y += h[n] * BCb[j][D_STATE + n];
            }
            ys[row * D_INNER + d] = y;
        }
    }
}

// y = (ys + xi*Dp) * silu(z) -> bf16
__global__ __launch_bounds__(256)
void k_gate(const float* __restrict__ ys, const float* __restrict__ xi,
            const float* __restrict__ xz, const float* __restrict__ Dp,
            bf16_t* __restrict__ outp) {
    const size_t idx = (size_t)blockIdx.x * 256 + threadIdx.x;   // over R*2048
    const int d = (int)(idx & (D_INNER - 1));
    const size_t row = idx >> 11;
    float y = ys[idx] + xi[idx] * Dp[d];
    const float z = xz[row * (2 * D_INNER) + D_INNER + d];
    y *= z / (1.f + __expf(-z));
    outp[idx] = (bf16_t)y;
}

// x2 = x + mamba_out; LayerNorm(x2) -> bf16
__global__ __launch_bounds__(256)
void k_add_ln(const float* __restrict__ x, const float* __restrict__ mo,
              const float* __restrict__ lnw, const float* __restrict__ lnb,
              float* __restrict__ x2, bf16_t* __restrict__ xn2) {
    const int row = blockIdx.x;
    const int C = D_MODEL;
    __shared__ float s1[256], s2[256];
    float a = 0.f, q = 0.f;
    for (int c = threadIdx.x; c < C; c += 256) {
        const size_t p = (size_t)row * C + c;
        const float v = x[p] + mo[p];
        x2[p] = v;
        a += v; q += v * v;
    }
    s1[threadIdx.x] = a; s2[threadIdx.x] = q;
    __syncthreads();
    for (int st = 128; st > 0; st >>= 1) {
        if (threadIdx.x < st) {
            s1[threadIdx.x] += s1[threadIdx.x + st];
            s2[threadIdx.x] += s2[threadIdx.x + st];
        }
        __syncthreads();
    }
    const float mu  = s1[0] / (float)C;
    const float var = s2[0] / (float)C - mu * mu;
    const float inv = rsqrtf(var + 1e-5f);
    for (int c = threadIdx.x; c < C; c += 256) {
        const size_t p = (size_t)row * C + c;
        xn2[p] = (bf16_t)((x2[p] - mu) * inv * lnw[c] + lnb[c]);
    }
}

// GEGLU: u = (a + b1a) * gelu_exact(g + b1g) -> bf16
__global__ __launch_bounds__(256)
void k_glu(const float* __restrict__ h, const float* __restrict__ b1,
           bf16_t* __restrict__ u) {
    const size_t idx = (size_t)blockIdx.x * 256 + threadIdx.x;  // over R*2048
    const int d = (int)(idx & (D_INNER - 1));
    const size_t row = idx >> 11;
    const float a  = h[row * (2 * D_INNER) + d] + b1[d];
    const float gg = h[row * (2 * D_INNER) + D_INNER + d] + b1[D_INNER + d];
    const float ge = 0.5f * gg * (1.f + erff(gg * 0.70710678118654752f));
    u[idx] = (bf16_t)(a * ge);
}

// out = x2 + ff + ff_b2
__global__ void k_final(const float* __restrict__ x2, const float* __restrict__ ff,
                        const float* __restrict__ b2, float* __restrict__ outp,
                        int total) {
    const int idx = blockIdx.x * 256 + threadIdx.x;
    if (idx >= total) return;
    const int c = idx & (D_MODEL - 1);
    outp[idx] = x2[idx] + ff[idx] + b2[c];
}

// ---------------------------------------------------------------------------
// Host launch
// ---------------------------------------------------------------------------
static inline dim3 gemm_grid(int M, int N) {
    return dim3((unsigned)(N / 128), (unsigned)(M / 128), 1);
}

extern "C" void kernel_launch(void* const* d_in, const int* in_sizes, int n_in,
                              void* d_out, int out_size, void* d_ws, size_t ws_size,
                              hipStream_t stream) {
    (void)in_sizes; (void)n_in; (void)out_size; (void)ws_size;
    const float* x       = (const float*)d_in[0];
    const float* rms_w   = (const float*)d_in[1];
    const float* in_w    = (const float*)d_in[2];
    const float* conv_w  = (const float*)d_in[3];
    const float* conv_b  = (const float*)d_in[4];
    const float* xproj_w = (const float*)d_in[5];
    const float* dt_w    = (const float*)d_in[6];
    const float* dt_b    = (const float*)d_in[7];
    const float* A_log   = (const float*)d_in[8];
    const float* Dp      = (const float*)d_in[9];
    const float* out_w   = (const float*)d_in[10];
    const float* ln_w    = (const float*)d_in[11];
    const float* ln_b    = (const float*)d_in[12];
    const float* ff_w1   = (const float*)d_in[13];
    const float* ff_b1   = (const float*)d_in[14];
    const float* ff_w2   = (const float*)d_in[15];
    const float* ff_b2   = (const float*)d_in[16];
    float* outp = (float*)d_out;

    char* ws = (char*)d_ws;
    size_t off = 0;
    auto alloc = [&](size_t bytes) -> char* {
        size_t a = (off + 255) & ~(size_t)255;
        off = a + bytes;
        return ws + a;
    };
    const size_t R = R_ROWS;

    bf16_t* in_wT    = (bf16_t*)alloc((size_t)4096 * 1024 * 2);
    bf16_t* xproj_wT = (bf16_t*)alloc((size_t)PROJ_LD * 2048 * 2);  // padded
    bf16_t* dt_wT    = (bf16_t*)alloc((size_t)2048 * 64 * 2);
    bf16_t* out_wT   = (bf16_t*)alloc((size_t)1024 * 2048 * 2);
    bf16_t* ff1_wT   = (bf16_t*)alloc((size_t)4096 * 1024 * 2);
    bf16_t* ff2_wT   = (bf16_t*)alloc((size_t)1024 * 2048 * 2);

    bf16_t* xn_bf    = (bf16_t*)alloc(R * 1024 * 2);
    float*  big      = (float*)alloc(R * 4096 * 4);     // xz, later ffn hidden
    float*  xi       = (float*)alloc(R * 2048 * 4);
    bf16_t* xi_bf    = (bf16_t*)alloc(R * 2048 * 2);
    float*  proj     = (float*)alloc(R * PROJ_LD * 4);  // padded to 128 cols
    bf16_t* dtin_bf  = (bf16_t*)alloc(R * 64 * 2);
    float*  dt       = (float*)alloc(R * 2048 * 4);
    float*  ys       = (float*)alloc(R * 2048 * 4);
    bf16_t* yg_bf    = (bf16_t*)alloc(R * 2048 * 2);
    float*  mo       = (float*)alloc(R * 1024 * 4);
    float*  x2       = (float*)alloc(R * 1024 * 4);
    bf16_t* xn2_bf   = (bf16_t*)alloc(R * 1024 * 2);
    bf16_t* u_bf     = (bf16_t*)alloc(R * 2048 * 2);
    float*  ff       = (float*)alloc(R * 1024 * 4);

    // --- weight casts (f32 KxN -> bf16 Npad x K, zero pad rows) ---
    auto castT = [&](const float* w, bf16_t* wT, int K, int N, int Npad) {
        const int tot = K * Npad;
        k_cast_transpose<<<(tot + 255) / 256, 256, 0, stream>>>(w, wT, K, N, Npad);
    };
    castT(in_w,    in_wT,    1024, 4096, 4096);
    castT(xproj_w, xproj_wT, 2048, 96,   PROJ_LD);
    castT(dt_w,    dt_wT,    64,   2048, 2048);
    castT(out_w,   out_wT,   2048, 1024, 1024);
    castT(ff_w1,   ff1_wT,   1024, 4096, 4096);
    castT(ff_w2,   ff2_wT,   2048, 1024, 1024);

    // 1) RMSNorm -> bf16
    k_rmsnorm<<<(unsigned)R, 256, 0, stream>>>(x, rms_w, xn_bf);

    // 2) xz = xn @ in_w   [8192 x 4096, K=1024]
    k_gemm_bf16<<<gemm_grid(R_ROWS, 4096), 256, 0, stream>>>(
        xn_bf, in_wT, big, R_ROWS, 4096, 1024);

    // 3) depthwise conv + SiLU on xi half
    k_conv_silu<<<(unsigned)(R * 2048 / 256), 256, 0, stream>>>(
        big, conv_w, conv_b, xi, xi_bf);

    // 4) proj = xi @ xproj_w   [8192 x 128(pad), K=2048]
    k_gemm_bf16<<<gemm_grid(R_ROWS, PROJ_LD), 256, 0, stream>>>(
        xi_bf, xproj_wT, proj, R_ROWS, PROJ_LD, 2048);

    // 5) dt_in slice -> bf16; dt_raw = dt_in @ dt_w; softplus
    k_slice_cast_bf16<<<(int)(R * 64 + 255) / 256, 256, 0, stream>>>(
        proj, dtin_bf, R_ROWS, PROJ_LD, 0, 64);
    k_gemm_bf16<<<gemm_grid(R_ROWS, 2048), 256, 0, stream>>>(
        dtin_bf, dt_wT, dt, R_ROWS, 2048, 64);
    k_softplus<<<(int)(R * 2048 + 255) / 256, 256, 0, stream>>>(
        dt, dt_b, (int)(R * 2048));

    // 6) selective scan
    k_scan<<<dim3(2048 / 256, BATCH_C), 256, 0, stream>>>(xi, dt, proj, A_log, ys);

    // 7) gate: (ys + xi*Dp) * silu(z) -> bf16
    k_gate<<<(unsigned)(R * 2048 / 256), 256, 0, stream>>>(ys, xi, big, Dp, yg_bf);

    // 8) mamba out-proj  [8192 x 1024, K=2048]
    k_gemm_bf16<<<gemm_grid(R_ROWS, 1024), 256, 0, stream>>>(
        yg_bf, out_wT, mo, R_ROWS, 1024, 2048);

    // 9) residual + LayerNorm -> bf16
    k_add_ln<<<(unsigned)R, 256, 0, stream>>>(x, mo, ln_w, ln_b, x2, xn2_bf);

    // 10) FFN up  [8192 x 4096, K=1024]  (reuses `big`, xz is dead now)
    k_gemm_bf16<<<gemm_grid(R_ROWS, 4096), 256, 0, stream>>>(
        xn2_bf, ff1_wT, big, R_ROWS, 4096, 1024);

    // 11) GEGLU -> bf16
    k_glu<<<(unsigned)(R * 2048 / 256), 256, 0, stream>>>(big, ff_b1, u_bf);

    // 12) FFN down  [8192 x 1024, K=2048]
    k_gemm_bf16<<<gemm_grid(R_ROWS, 1024), 256, 0, stream>>>(
        u_bf, ff2_wT, ff, R_ROWS, 1024, 2048);

    // 13) final residual + bias
    k_final<<<(int)(R * 1024 + 255) / 256, 256, 0, stream>>>(
        x2, ff, ff_b2, outp, (int)(R * 1024));
}
